// LabelContextAttentionBlock_60739427500845
// MI455X (gfx1250) — compile-verified, matrix operand order checked by reading, not run on hardware
//
#include <hip/hip_runtime.h>
#include <math.h>

// ---------------------------------------------------------------------------
// LabelContextAttentionBlock for MI455X (gfx1250, wave32, WMMA).
// All GEMMs via v_wmma_f32_16x16x32_bf16 (fp32 -> bf16 at LDS staging, f32
// accumulate). Staging is float4 vectorized AND software-pipelined: global
// loads for chunk k+1 are issued before the WMMA phase of chunk k so HBM
// latency overlaps matrix math. Context GEMM is K-split 4x with a
// deterministic partial-sum reduction (no atomics).
// ---------------------------------------------------------------------------

typedef __attribute__((ext_vector_type(16))) __bf16 v16bf;
typedef __attribute__((ext_vector_type(4)))  __bf16 v4bf;
typedef __attribute__((ext_vector_type(8)))  float  v8f;

#define BATCH 8
#define CIN   512
#define OCH   256
#define LBL   19
#define LP    32
#define HWTOT 9216   // 96*96
#define EPS   1e-5f
#define KSPLIT 4
#define CTXN  (BATCH * CIN * LBL)   // 77824

__device__ __forceinline__ v8f zero8() {
  v8f z = {0.f, 0.f, 0.f, 0.f, 0.f, 0.f, 0.f, 0.f};
  return z;
}

// 4 floats (in regs) -> 4x bf16 packed LDS store (cvt_pk pair + ds_store_b64)
__device__ __forceinline__ void put4(__bf16* dst, float4 v) {
  v4bf o;
  o[0] = (__bf16)v.x; o[1] = (__bf16)v.y; o[2] = (__bf16)v.z; o[3] = (__bf16)v.w;
  *(v4bf*)dst = o;
}

// 16x32 bf16 fragment from LDS, row-major [row][K], row stride `rstride`.
// Per-lane layout (ISA 7.12.2, wave32, 16-bit A; also serves B stored [N][K]):
//   lanes 0-15 : row = lane,    K = 0..7  and 16..23
//   lanes 16-31: row = lane-16, K = 8..15 and 24..31
__device__ __forceinline__ v16bf load_frag(const __bf16* p, int rstride) {
  const int lane = threadIdx.x & 31;
  const __bf16* row = p + (lane & 15) * rstride + ((lane >> 4) << 3);
  v16bf f;
#pragma unroll
  for (int i = 0; i < 8; ++i) f[i] = row[i];
#pragma unroll
  for (int i = 0; i < 8; ++i) f[i + 8] = row[16 + i];
  return f;
}

__device__ __forceinline__ v8f wmma_bf16(v16bf a, v16bf b, v8f c) {
  return __builtin_amdgcn_wmma_f32_16x16x32_bf16(false, a, false, b,
                                                 (short)0, c, false, false);
}

// ---------------------------------------------------------------------------
// Kernel 1: partial context: ctxp[kz][b,c,l] = sum_{h slice} P[b,l,h]*S[b,c,h]
// Grid (8, 8, KSPLIT), 256 threads (8 waves: 4 M-tiles x 2 N-tiles).
// Software-pipelined: regs hold chunk k+1 while chunk k is in LDS/WMMA.
// ---------------------------------------------------------------------------
__global__ __launch_bounds__(256)
void ctx_kernel(const float* __restrict__ S, const float* __restrict__ P,
                float* __restrict__ ctxp) {
  __shared__ __attribute__((aligned(16))) __bf16 sS[64 * 32];  // A: [c][h]
  __shared__ __attribute__((aligned(16))) __bf16 sP[32 * 32];  // B^T: [l][h]

  const int tid  = threadIdx.x;
  const int lane = tid & 31;
  const int wave = tid >> 5;
  const int hi8  = (lane >> 4) << 3;
  const int col  = lane & 15;
  const int b  = blockIdx.x;
  const int c0 = blockIdx.y * 64;
  const int kz = blockIdx.z;
  const int m = wave >> 1;     // 0..3
  const int n = wave & 1;      // 0..1

  const float* Sb = S + ((long)b * CIN + c0) * HWTOT;
  const float* Pb = P + (long)b * LBL * HWTOT;

  // per-thread staging coordinates (2 quads of S, <=1 quad of P)
  const int sc0 = tid >> 3,        sh0 = (tid & 7) * 4;          // quad 0
  const int sc1 = (tid + 256) >> 3, sh1 = ((tid + 256) & 7) * 4; // quad 1
  const bool hasP = tid < LBL * 8;
  const int pl = tid >> 3, ph = (tid & 7) * 4;

  // zero the l-padding rows of sP once (rows 19..31 never rewritten)
  for (int i = LBL * 32 + tid; i < 32 * 32; i += 256) sP[i] = (__bf16)0.f;

  const int ks0 = kz * (HWTOT / 32 / KSPLIT);
  const int ks1 = ks0 + (HWTOT / 32 / KSPLIT);

  float4 rs0, rs1, rp;
  {  // preload chunk ks0
    const int h0 = ks0 * 32;
    rs0 = *(const float4*)(Sb + (long)sc0 * HWTOT + h0 + sh0);
    rs1 = *(const float4*)(Sb + (long)sc1 * HWTOT + h0 + sh1);
    if (hasP) rp = *(const float4*)(Pb + (long)pl * HWTOT + h0 + ph);
  }

  v8f acc = zero8();
  for (int ks = ks0; ks < ks1; ++ks) {
    // commit regs -> LDS
    put4(sS + sc0 * 32 + sh0, rs0);
    put4(sS + sc1 * 32 + sh1, rs1);
    if (hasP) put4(sP + pl * 32 + ph, rp);
    __syncthreads();
    // issue global loads for next chunk; they overlap the WMMA below
    if (ks + 1 < ks1) {
      const int h0 = (ks + 1) * 32;
      rs0 = *(const float4*)(Sb + (long)sc0 * HWTOT + h0 + sh0);
      rs1 = *(const float4*)(Sb + (long)sc1 * HWTOT + h0 + sh1);
      if (hasP) rp = *(const float4*)(Pb + (long)pl * HWTOT + h0 + ph);
      __builtin_prefetch(Sb + (long)sc0 * HWTOT + h0 + 32 + sh0, 0, 3);
    }
    v16bf a  = load_frag(sS + m * 16 * 32, 32);
    v16bf bb = load_frag(sP + n * 16 * 32, 32);
    acc = wmma_bf16(a, bb, acc);
    __syncthreads();
  }
#pragma unroll
  for (int r = 0; r < 8; ++r) {
    const int c = c0 + m * 16 + r + hi8;
    const int l = n * 16 + col;
    if (l < LBL)
      ctxp[(long)kz * CTXN + ((long)b * CIN + c) * LBL + l] = acc[r];
  }
}

// Deterministic partial-sum reduction: ctxp[0][i] = sum_kz ctxp[kz][i]
__global__ __launch_bounds__(256)
void ctx_reduce_kernel(float* __restrict__ ctxp) {
  const int i = blockIdx.x * 256 + threadIdx.x;
  if (i < CTXN)
    ctxp[i] = ctxp[i] + ctxp[CTXN + i] + ctxp[2 * CTXN + i] + ctxp[3 * CTXN + i];
}

// ---------------------------------------------------------------------------
// Kernel 2: key / value path (tiny ~0.1 GF). Scalar FMA, ctx cached in LDS.
// Grid (8), 256 threads (thread = output channel o).
// ---------------------------------------------------------------------------
__global__ __launch_bounds__(256)
void kv_kernel(const float* __restrict__ ctx,
               const float* __restrict__ Wk1, const float* __restrict__ bnk1,
               const float* __restrict__ Wk2, const float* __restrict__ bnk2,
               const float* __restrict__ Wv,  const float* __restrict__ bnv,
               float* __restrict__ keyb, float* __restrict__ valb) {
  __shared__ __attribute__((aligned(16))) float ctxL[CIN * LBL];  // 38.9 KB
  __shared__ float k1L[OCH * LBL];                                // 19.4 KB

  const int tid = threadIdx.x;
  const int b   = blockIdx.x;
  for (int i = tid; i < CIN * LBL / 4; i += 256)
    *(float4*)&ctxL[i * 4] = *(const float4*)&ctx[(long)b * CIN * LBL + i * 4];
  __syncthreads();

  const int o = tid;
  float aK[LBL], aV[LBL];
#pragma unroll
  for (int l = 0; l < LBL; ++l) { aK[l] = 0.f; aV[l] = 0.f; }
  for (int c = 0; c < CIN; ++c) {
    const float w1 = Wk1[o * CIN + c];
    const float wv = Wv[o * CIN + c];
    const float* cr = &ctxL[c * LBL];
#pragma unroll
    for (int l = 0; l < LBL; ++l) {
      aK[l] = fmaf(w1, cr[l], aK[l]);
      aV[l] = fmaf(wv, cr[l], aV[l]);
    }
  }
  {
    const float s  = bnk1[o] * rsqrtf(bnk1[3 * OCH + o] + EPS);
    const float sh = bnk1[OCH + o] - bnk1[2 * OCH + o] * s;
#pragma unroll
    for (int l = 0; l < LBL; ++l) k1L[o * LBL + l] = fmaxf(aK[l] * s + sh, 0.f);
  }
  {
    const float s  = bnv[o] * rsqrtf(bnv[3 * OCH + o] + EPS);
    const float sh = bnv[OCH + o] - bnv[2 * OCH + o] * s;
#pragma unroll
    for (int l = 0; l < LBL; ++l)
      valb[((long)b * OCH + o) * LBL + l] = fmaxf(aV[l] * s + sh, 0.f);
  }
  __syncthreads();

  float aK2[LBL];
#pragma unroll
  for (int l = 0; l < LBL; ++l) aK2[l] = 0.f;
  for (int p = 0; p < OCH; ++p) {
    const float w2 = Wk2[o * OCH + p];
    const float* cr = &k1L[p * LBL];
#pragma unroll
    for (int l = 0; l < LBL; ++l) aK2[l] = fmaf(w2, cr[l], aK2[l]);
  }
  const float s2  = bnk2[o] * rsqrtf(bnk2[3 * OCH + o] + EPS);
  const float sh2 = bnk2[OCH + o] - bnk2[2 * OCH + o] * s2;
#pragma unroll
  for (int l = 0; l < LBL; ++l)
    keyb[((long)b * OCH + o) * LBL + l] = fmaxf(aK2[l] * s2 + sh2, 0.f);
}

// ---------------------------------------------------------------------------
// Kernel 3: fused query path + attention, per (batch, 64-pixel tile).
//   GEMM1: q1 = relu(bn(Wq1 @ X))      256x64, K=512  (pipelined staging)
//   GEMM2: q2 = relu(bn(Wq2 @ q1))     256x64, K=256  (q1 in LDS)
//   GEMM3: sim = (q2^T @ key)/16       64x32,  K=256
//   softmax over L=19
//   GEMM4: out = value @ sim^T         256x64, K=32 (zero-padded)
// Grid (144, 8), 256 threads. LDS overlaid: 56 KB.
// ---------------------------------------------------------------------------
__global__ __launch_bounds__(256)
void attn_kernel(const float* __restrict__ X,
                 const float* __restrict__ Wq1, const float* __restrict__ bnq1,
                 const float* __restrict__ Wq2, const float* __restrict__ bnq2,
                 const float* __restrict__ keyb, const float* __restrict__ valb,
                 float* __restrict__ out) {
  __shared__ __attribute__((aligned(16))) __bf16 sA[256 * 32];   // 16 KB: W-chunk, later key^T [l][o]
  __shared__ __attribute__((aligned(16))) __bf16 sXT[64 * 32];   // 4 KB : X^T chunk [hw][c]
  __shared__ __attribute__((aligned(16))) unsigned char sBig[32768];
  // sBig phase 1: qbuf = q1^T / q2^T bf16 [hw][o]  (32 KB)
  // sBig phase 2: vA [o][32] bf16 (16 KB) + simf f32 [hw][32] (8 KB) + simb bf16 (4 KB)
  __shared__ float sSc1[OCH], sSh1[OCH], sSc2[OCH], sSh2[OCH];

  const int tid  = threadIdx.x;
  const int lane = tid & 31;
  const int wave = tid >> 5;
  const int hi8  = (lane >> 4) << 3;
  const int col  = lane & 15;
  const int b    = blockIdx.y;
  const int hw0  = blockIdx.x * 64;

  __bf16* qbuf = (__bf16*)sBig;
  const float* Xb = X + (long)b * CIN * HWTOT + hw0;

  // staging coordinates: 8 W-quads and 2 X-quads per thread
  const int xq0c = tid >> 4,         xq0h = (tid & 15) * 4;
  const int xq1c = (tid + 256) >> 4, xq1h = ((tid + 256) & 15) * 4;

  {  // fold BN params: y = acc*sc + sh
    const int o = tid;
    float s = bnq1[o] * rsqrtf(bnq1[3 * OCH + o] + EPS);
    sSc1[o] = s; sSh1[o] = bnq1[OCH + o] - bnq1[2 * OCH + o] * s;
    s = bnq2[o] * rsqrtf(bnq2[3 * OCH + o] + EPS);
    sSc2[o] = s; sSh2[o] = bnq2[OCH + o] - bnq2[2 * OCH + o] * s;
  }

  // ---- GEMM1: wave w owns M-tiles {2w,2w+1} x N-tiles {0..3}
  v8f acc1[2][4];
#pragma unroll
  for (int mi = 0; mi < 2; ++mi)
#pragma unroll
    for (int ni = 0; ni < 4; ++ni) acc1[mi][ni] = zero8();

  float4 rw[8], rx0, rx1;
#pragma unroll
  for (int j = 0; j < 8; ++j) {   // preload Wq1 chunk 0
    const int q = tid + j * 256;
    rw[j] = *(const float4*)(Wq1 + (q >> 3) * CIN + (q & 7) * 4);
  }
  rx0 = *(const float4*)(Xb + (long)xq0c * HWTOT + xq0h);
  rx1 = *(const float4*)(Xb + (long)xq1c * HWTOT + xq1h);

  for (int ks = 0; ks < CIN / 32; ++ks) {
    // commit regs -> LDS
#pragma unroll
    for (int j = 0; j < 8; ++j) {
      const int q = tid + j * 256;
      put4(sA + (q >> 3) * 32 + (q & 7) * 4, rw[j]);
    }
    sXT[(xq0h + 0) * 32 + xq0c] = (__bf16)rx0.x;
    sXT[(xq0h + 1) * 32 + xq0c] = (__bf16)rx0.y;
    sXT[(xq0h + 2) * 32 + xq0c] = (__bf16)rx0.z;
    sXT[(xq0h + 3) * 32 + xq0c] = (__bf16)rx0.w;
    sXT[(xq1h + 0) * 32 + xq1c] = (__bf16)rx1.x;
    sXT[(xq1h + 1) * 32 + xq1c] = (__bf16)rx1.y;
    sXT[(xq1h + 2) * 32 + xq1c] = (__bf16)rx1.z;
    sXT[(xq1h + 3) * 32 + xq1c] = (__bf16)rx1.w;
    __syncthreads();
    // issue next-chunk loads; they overlap the WMMA below
    if (ks + 1 < CIN / 32) {
      const int kc = (ks + 1) * 32;
#pragma unroll
      for (int j = 0; j < 8; ++j) {
        const int q = tid + j * 256;
        rw[j] = *(const float4*)(Wq1 + (q >> 3) * CIN + kc + (q & 7) * 4);
      }
      rx0 = *(const float4*)(Xb + (long)(kc + xq0c) * HWTOT + xq0h);
      rx1 = *(const float4*)(Xb + (long)(kc + xq1c) * HWTOT + xq1h);
      __builtin_prefetch(Xb + (long)(kc + 32 + xq0c) * HWTOT + xq0h, 0, 3);
    }
#pragma unroll
    for (int mi = 0; mi < 2; ++mi) {
      v16bf a = load_frag(sA + (wave * 2 + mi) * 16 * 32, 32);
#pragma unroll
      for (int ni = 0; ni < 4; ++ni) {
        v16bf bb = load_frag(sXT + ni * 16 * 32, 32);
        acc1[mi][ni] = wmma_bf16(a, bb, acc1[mi][ni]);
      }
    }
    __syncthreads();
  }
  // BN1 + ReLU, transpose into qbuf[hw][o]
#pragma unroll
  for (int mi = 0; mi < 2; ++mi)
#pragma unroll
    for (int ni = 0; ni < 4; ++ni)
#pragma unroll
      for (int r = 0; r < 8; ++r) {
        const int o  = (wave * 2 + mi) * 16 + r + hi8;
        const int hw = ni * 16 + col;
        const float y = acc1[mi][ni][r] * sSc1[o] + sSh1[o];
        qbuf[hw * OCH + o] = (__bf16)fmaxf(y, 0.f);
      }

  // ---- GEMM2 (pipelined Wq2 staging)
  v8f acc2[2][4];
#pragma unroll
  for (int mi = 0; mi < 2; ++mi)
#pragma unroll
    for (int ni = 0; ni < 4; ++ni) acc2[mi][ni] = zero8();

#pragma unroll
  for (int j = 0; j < 8; ++j) {
    const int q = tid + j * 256;
    rw[j] = *(const float4*)(Wq2 + (q >> 3) * OCH + (q & 7) * 4);
  }
  for (int ks = 0; ks < OCH / 32; ++ks) {
#pragma unroll
    for (int j = 0; j < 8; ++j) {
      const int q = tid + j * 256;
      put4(sA + (q >> 3) * 32 + (q & 7) * 4, rw[j]);
    }
    __syncthreads();
    if (ks + 1 < OCH / 32) {
      const int kc = (ks + 1) * 32;
#pragma unroll
      for (int j = 0; j < 8; ++j) {
        const int q = tid + j * 256;
        rw[j] = *(const float4*)(Wq2 + (q >> 3) * OCH + kc + (q & 7) * 4);
      }
    }
    const int kc = ks * 32;
#pragma unroll
    for (int mi = 0; mi < 2; ++mi) {
      v16bf a = load_frag(sA + (wave * 2 + mi) * 16 * 32, 32);
#pragma unroll
      for (int ni = 0; ni < 4; ++ni) {
        v16bf bb = load_frag(qbuf + ni * 16 * OCH + kc, OCH);
        acc2[mi][ni] = wmma_bf16(a, bb, acc2[mi][ni]);
      }
    }
    __syncthreads();
  }
  // BN2 + ReLU -> overwrite qbuf with q2^T; stage key^T [l][o] into sA
#pragma unroll
  for (int mi = 0; mi < 2; ++mi)
#pragma unroll
    for (int ni = 0; ni < 4; ++ni)
#pragma unroll
      for (int r = 0; r < 8; ++r) {
        const int o  = (wave * 2 + mi) * 16 + r + hi8;
        const int hw = ni * 16 + col;
        const float y = acc2[mi][ni][r] * sSc2[o] + sSh2[o];
        qbuf[hw * OCH + o] = (__bf16)fmaxf(y, 0.f);
      }
  for (int i = tid; i < LP * OCH; i += 256) {
    const int l = i >> 8, o = i & 255;
    sA[i] = (l < LBL) ? (__bf16)keyb[((long)b * OCH + o) * LBL + l] : (__bf16)0.f;
  }
  __syncthreads();

  // ---- GEMM3: sim = q2^T @ key * (1/16); 8 tiles, one per wave
  v8f acc3 = zero8();
  {
    const int m = wave >> 1, n = wave & 1;
    for (int ks = 0; ks < OCH / 32; ++ks) {
      v16bf a  = load_frag(qbuf + m * 16 * OCH + ks * 32, OCH);
      v16bf bb = load_frag(sA + n * 16 * OCH + ks * 32, OCH);
      acc3 = wmma_bf16(a, bb, acc3);
    }
  }
  __syncthreads();   // qbuf dead: reuse region

  __bf16* vA   = (__bf16*)sBig;             // [o][32], l>=19 zero
  float*  simf = (float*)(sBig + 16384);    // [hw][32]
  __bf16* simb = (__bf16*)(sBig + 24576);   // [hw][32], softmaxed

  {
    const int m = wave >> 1, n = wave & 1;
#pragma unroll
    for (int r = 0; r < 8; ++r) {
      const int hw = m * 16 + r + hi8;
      const int l  = n * 16 + col;
      simf[hw * 32 + l] = acc3[r] * 0.0625f;   // 1/sqrt(256)
    }
  }
  {  // value rows -> vA[o][32] with zero l-padding (one row per thread)
    const float* vr = valb + ((long)b * OCH + tid) * LBL;
    __bf16* dst = vA + tid * LP;
#pragma unroll
    for (int l = 0; l < LBL; ++l) dst[l] = (__bf16)vr[l];
#pragma unroll
    for (int l = LBL; l < LP; ++l) dst[l] = (__bf16)0.f;
  }
  __syncthreads();

  // ---- softmax over l (19) per pixel row
  if (tid < 64) {
    const float* row = simf + tid * 32;
    float mx = -3.4e38f;
#pragma unroll
    for (int l = 0; l < LBL; ++l) mx = fmaxf(mx, row[l]);
    float e[LBL], s = 0.f;
#pragma unroll
    for (int l = 0; l < LBL; ++l) { e[l] = __expf(row[l] - mx); s += e[l]; }
    const float inv = 1.f / s;
    __bf16* rb = simb + tid * 32;
#pragma unroll
    for (int l = 0; l < LBL; ++l) rb[l] = (__bf16)(e[l] * inv);
#pragma unroll
    for (int l = LBL; l < LP; ++l) rb[l] = (__bf16)0.f;
  }
  __syncthreads();

  // ---- GEMM4: out = value @ sim^T, single K=32 step; store f32 to HBM
  const long obase = (long)b * OCH * HWTOT + hw0;
#pragma unroll
  for (int mi = 0; mi < 2; ++mi) {
    v16bf a = load_frag(vA + (wave * 2 + mi) * 16 * 32, 32);
#pragma unroll
    for (int ni = 0; ni < 4; ++ni) {
      v16bf bb = load_frag(simb + ni * 16 * 32, 32);
      v8f o4 = wmma_bf16(a, bb, zero8());
#pragma unroll
      for (int r = 0; r < 8; ++r) {
        const int o  = (wave * 2 + mi) * 16 + r + hi8;
        const int hw = ni * 16 + col;
        out[obase + (long)o * HWTOT + hw] = o4[r];
      }
    }
  }
}

// ---------------------------------------------------------------------------
extern "C" void kernel_launch(void* const* d_in, const int* in_sizes, int n_in,
                              void* d_out, int out_size, void* d_ws, size_t ws_size,
                              hipStream_t stream) {
  const float* X    = (const float*)d_in[0];   // target_task_feats [8,512,9216]
  const float* S    = (const float*)d_in[1];   // source_task_feats [8,512,9216]
  const float* P    = (const float*)d_in[2];   // target_aux_prob   [8,19,9216]
  // d_in[3] (source_aux_prob) unused by the reference
  const float* Wq1  = (const float*)d_in[4];
  const float* bnq1 = (const float*)d_in[5];
  const float* Wq2  = (const float*)d_in[6];
  const float* bnq2 = (const float*)d_in[7];
  const float* Wk1  = (const float*)d_in[8];
  const float* bnk1 = (const float*)d_in[9];
  const float* Wk2  = (const float*)d_in[10];
  const float* bnk2 = (const float*)d_in[11];
  const float* Wv   = (const float*)d_in[12];
  const float* bnv  = (const float*)d_in[13];
  float* out = (float*)d_out;

  // workspace: ctx partials [KSPLIT][8*512*19] (final sum lands in slice 0)
  //          + key [8,256,19] + value [8,256,19]   (~1.5 MB total)
  float* ctxp = (float*)d_ws;
  float* keyb = ctxp + (long)KSPLIT * CTXN;
  float* valb = keyb + (long)BATCH * OCH * LBL;

  ctx_kernel<<<dim3(BATCH, CIN / 64, KSPLIT), 256, 0, stream>>>(S, P, ctxp);
  ctx_reduce_kernel<<<dim3((CTXN + 255) / 256), 256, 0, stream>>>(ctxp);
  kv_kernel<<<dim3(BATCH), 256, 0, stream>>>(ctxp, Wk1, bnk1, Wk2, bnk2,
                                             Wv, bnv, keyb, valb);
  attn_kernel<<<dim3(HWTOT / 64, BATCH), 256, 0, stream>>>(
      X, Wq1, bnq1, Wq2, bnq2, keyb, valb, out);
}
